// SimpleSOLOV2_63891933495889
// MI455X (gfx1250) — matrix operand order repaired and checked by reading, not compiled, not run on hardware
//
#include <hip/hip_runtime.h>
#include <hip/hip_bf16.h>
#include <math.h>

typedef __attribute__((ext_vector_type(16))) _Float16 v16h;
typedef __attribute__((ext_vector_type(8)))  float    v8f;

#define SIGMA 2.0f

// ---------------------------------------------------------------------------
// 1) Per-row mask area: one block per row, float4 strided loads, LDS reduce.
// ---------------------------------------------------------------------------
__global__ void area_kernel(const float* __restrict__ masks,
                            float* __restrict__ area, int HW) {
    const int row = blockIdx.x;
    const float4* p4 = (const float4*)(masks + (size_t)row * HW);
    const int nv = HW >> 2;
    float s = 0.0f;
    for (int i = threadIdx.x; i < nv; i += blockDim.x) {
        float4 v = p4[i];
        s += v.x + v.y + v.z + v.w;
    }
    __shared__ float sh[256];
    sh[threadIdx.x] = s;
    __syncthreads();
    for (int off = 128; off > 0; off >>= 1) {
        if (threadIdx.x < off) sh[threadIdx.x] += sh[threadIdx.x + off];
        __syncthreads();
    }
    if (threadIdx.x == 0) area[row] = sh[0];
}

// ---------------------------------------------------------------------------
// 2) Stable descending rank-sort by score (N=200, one block). Scatters
//    order / scores_s / labels_s / area_s. Matches jnp.argsort(-scores).
// ---------------------------------------------------------------------------
__global__ void sort1_kernel(const float* __restrict__ scores,
                             const int* __restrict__ labels,
                             const float* __restrict__ area,
                             int* __restrict__ order,
                             float* __restrict__ scores_s,
                             int* __restrict__ labels_s,
                             float* __restrict__ area_s, int N) {
    __shared__ float s[256];
    const int i = threadIdx.x;
    if (i < N) s[i] = scores[i];
    __syncthreads();
    if (i < N) {
        const float si = s[i];
        int rank = 0;
        for (int j = 0; j < N; ++j) {
            const float sj = s[j];
            rank += (sj > si) || (sj == si && j < i);
        }
        order[rank]    = i;
        scores_s[rank] = si;
        labels_s[rank] = labels[i];
        area_s[rank]   = area[i];
    }
}

// ---------------------------------------------------------------------------
// Load one 32-wide K slab for A and B operands (fp32 -> f16 in registers),
// laid out per the CDNA5 ISA f16 operand maps:
//   A (16x32): lanes 0-15 hold K {k..k+7, k+16..k+23},
//              lanes 16-31 hold K {k+8..k+15, k+24..k+31}; M = lane&15
//   B (32x16): lanes 0-15 hold K {k..k+7, k+16..k+23},
//              lanes 16-31 hold K {k+8..k+15, k+24..k+31}; N = lane&15
// ---------------------------------------------------------------------------
__device__ __forceinline__ void load_ab(const float* __restrict__ Arow,
                                        const float* __restrict__ Brow,
                                        int k, int half, v16h& a, v16h& b) {
    const int ka0 = k + (half << 3);
    const int ka1 = k + 16 + (half << 3);
    float4 a0 = *(const float4*)(Arow + ka0);
    float4 a1 = *(const float4*)(Arow + ka0 + 4);
    float4 a2 = *(const float4*)(Arow + ka1);
    float4 a3 = *(const float4*)(Arow + ka1 + 4);

    const int kb0 = k + (half << 3);
    const int kb1 = k + 16 + (half << 3);
    float4 b0 = *(const float4*)(Brow + kb0);
    float4 b1 = *(const float4*)(Brow + kb0 + 4);
    float4 b2 = *(const float4*)(Brow + kb1);
    float4 b3 = *(const float4*)(Brow + kb1 + 4);

    a[0]  = (_Float16)a0.x; a[1]  = (_Float16)a0.y;
    a[2]  = (_Float16)a0.z; a[3]  = (_Float16)a0.w;
    a[4]  = (_Float16)a1.x; a[5]  = (_Float16)a1.y;
    a[6]  = (_Float16)a1.z; a[7]  = (_Float16)a1.w;
    a[8]  = (_Float16)a2.x; a[9]  = (_Float16)a2.y;
    a[10] = (_Float16)a2.z; a[11] = (_Float16)a2.w;
    a[12] = (_Float16)a3.x; a[13] = (_Float16)a3.y;
    a[14] = (_Float16)a3.z; a[15] = (_Float16)a3.w;

    b[0]  = (_Float16)b0.x; b[1]  = (_Float16)b0.y;
    b[2]  = (_Float16)b0.z; b[3]  = (_Float16)b0.w;
    b[4]  = (_Float16)b1.x; b[5]  = (_Float16)b1.y;
    b[6]  = (_Float16)b1.z; b[7]  = (_Float16)b1.w;
    b[8]  = (_Float16)b2.x; b[9]  = (_Float16)b2.y;
    b[10] = (_Float16)b2.z; b[11] = (_Float16)b2.w;
    b[12] = (_Float16)b3.x; b[13] = (_Float16)b3.y;
    b[14] = (_Float16)b3.z; b[15] = (_Float16)b3.w;
}

// ---------------------------------------------------------------------------
// 3) Gram matrix inter = flat_s @ flat_s^T via v_wmma_f32_16x16x32_f16.
//    Grid (NT,NT) tiles of 16x16; block = 256 threads = 8 waves splitting K.
//    Out-of-range rows are CLAMPED to row N-1 (branch-free hot loop):
//    the resulting garbage lands only in padded inter cells that the decay
//    kernel never reads (it only touches i < j < N).
//    Two accumulators break the WMMA->WMMA RAW chain so back-to-back WMMAs
//    and the next slab's loads/converts can overlap.
// ---------------------------------------------------------------------------
__global__ void gemm_kernel(const float* __restrict__ masks,
                            const int* __restrict__ order,
                            float* __restrict__ inter,
                            int N, int HW, int NP) {
    const int tileJ = blockIdx.x;
    const int tileI = blockIdx.y;
    const int tid   = threadIdx.x;
    const int wave  = tid >> 5;
    const int lane  = tid & 31;
    const int half  = lane >> 4;
    const int m16   = lane & 15;

    int gi = tileI * 16 + m16; if (gi >= N) gi = N - 1;   // clamp: padded
    int gj = tileJ * 16 + m16; if (gj >= N) gj = N - 1;   // cells unused
    const float* Arow = masks + (size_t)order[gi] * HW;
    const float* Brow = masks + (size_t)order[gj] * HW;

    const int ksteps = HW >> 5;              // K-chunks of 32
    const int per    = (ksteps + 7) >> 3;    // split over 8 waves
    const int k0 = wave * per;
    const int k1 = (k0 + per < ksteps) ? (k0 + per) : ksteps;

    v8f c0 = {0.f, 0.f, 0.f, 0.f, 0.f, 0.f, 0.f, 0.f};
    v8f c1 = {0.f, 0.f, 0.f, 0.f, 0.f, 0.f, 0.f, 0.f};

    int ks = k0;
    for (; ks + 1 < k1; ks += 2) {
        v16h a, b;
        load_ab(Arow, Brow, ks << 5, half, a, b);
        c0 = __builtin_amdgcn_wmma_f32_16x16x32_f16(
                 false, a, false, b, (short)0, c0, false, false);
        v16h a2, b2;
        load_ab(Arow, Brow, (ks + 1) << 5, half, a2, b2);
        c1 = __builtin_amdgcn_wmma_f32_16x16x32_f16(
                 false, a2, false, b2, (short)0, c1, false, false);
    }
    if (ks < k1) {
        v16h a, b;
        load_ab(Arow, Brow, ks << 5, half, a, b);
        c0 = __builtin_amdgcn_wmma_f32_16x16x32_f16(
                 false, a, false, b, (short)0, c0, false, false);
    }
#pragma unroll
    for (int v = 0; v < 8; ++v) c0[v] += c1[v];

    // Cross-wave reduction of the 8 partial 16x16 accumulators through LDS.
    __shared__ float red[8 * 256];
#pragma unroll
    for (int v = 0; v < 8; ++v) red[wave * 256 + lane * 8 + v] = c0[v];
    __syncthreads();

    float sum = 0.0f;
#pragma unroll
    for (int w = 0; w < 8; ++w) sum += red[w * 256 + tid];

    // C/D layout: slot v, lane L -> M = v + 8*(L>=16), N = L&15.
    const int L = tid >> 3;
    const int v = tid & 7;
    const int i = tileI * 16 + v + ((L >> 4) << 3);
    const int j = tileJ * 16 + (L & 15);
    inter[(size_t)i * NP + j] = sum;
}

// ---------------------------------------------------------------------------
// 4) Matrix-NMS decay: compensate = max over rows of triu-masked IoU, then
//    decay_coef[j] = min_i exp(SIGMA*(comp[i]^2 - d_ij^2)). One block.
// ---------------------------------------------------------------------------
__global__ void decay_kernel(const float* __restrict__ inter,
                             const float* __restrict__ area_s,
                             const int* __restrict__ labels_s,
                             const float* __restrict__ scores_s,
                             float* __restrict__ new_scores, int N, int NP) {
    __shared__ float comp[256];
    __shared__ float ar[256];
    __shared__ int   lb[256];
    const int j = threadIdx.x;
    if (j < N) { ar[j] = area_s[j]; lb[j] = labels_s[j]; }
    __syncthreads();
    if (j < N) {
        const float aj = ar[j]; const int lj = lb[j];
        float cmax = 0.0f;
        for (int i = 0; i < N; ++i) {
            if (i < j && lb[i] == lj) {
                const float in  = inter[(size_t)i * NP + j];
                const float iou = in / (ar[i] + aj - in);
                cmax = fmaxf(cmax, iou);
            }
        }
        comp[j] = cmax;
    }
    __syncthreads();
    if (j < N) {
        const float aj = ar[j]; const int lj = lb[j];
        float coef = 3.4e38f;
        for (int i = 0; i < N; ++i) {
            float d = 0.0f;
            if (i < j && lb[i] == lj) {
                const float in = inter[(size_t)i * NP + j];
                d = in / (ar[i] + aj - in);
            }
            const float ci = comp[i];
            const float term = expf(SIGMA * (ci * ci - d * d));
            coef = fminf(coef, term);
        }
        new_scores[j] = scores_s[j] * coef;
    }
}

// ---------------------------------------------------------------------------
// 5) Second stable descending sort on new_scores; emit scalar outputs and
//    the composed permutation into the original mask rows.
// ---------------------------------------------------------------------------
__global__ void sort2_kernel(const float* __restrict__ new_scores,
                             const int* __restrict__ labels_s,
                             const int* __restrict__ order,
                             float* __restrict__ out_scores,
                             float* __restrict__ out_labels,
                             float* __restrict__ out_keep,
                             int* __restrict__ final_order, int N) {
    __shared__ float s[256];
    const int i = threadIdx.x;
    if (i < N) s[i] = new_scores[i];
    __syncthreads();
    if (i < N) {
        const float si = s[i];
        int rank = 0;
        for (int j = 0; j < N; ++j) {
            const float sj = s[j];
            rank += (sj > si) || (sj == si && j < i);
        }
        out_scores[rank]  = si;
        out_labels[rank]  = (float)labels_s[i];
        out_keep[rank]    = (float)order[i];   // keep_inds = order[order2]
        final_order[rank] = order[i];          // original mask row for slot
    }
}

// ---------------------------------------------------------------------------
// 6) Permuted mask copy (float4, bandwidth-bound).
// ---------------------------------------------------------------------------
__global__ void copy_masks_kernel(const float* __restrict__ masks,
                                  const int* __restrict__ final_order,
                                  float* __restrict__ out_masks, int HW) {
    const int r   = blockIdx.y;
    const int src = final_order[r];
    const float4* in  = (const float4*)(masks + (size_t)src * HW);
    float4*       out = (float4*)(out_masks + (size_t)r * HW);
    const int nv = HW >> 2;
    for (int i = blockIdx.x * blockDim.x + threadIdx.x; i < nv;
         i += gridDim.x * blockDim.x)
        out[i] = in[i];
}

// ---------------------------------------------------------------------------
extern "C" void kernel_launch(void* const* d_in, const int* in_sizes, int n_in,
                              void* d_out, int out_size, void* d_ws, size_t ws_size,
                              hipStream_t stream) {
    const float* masks  = (const float*)d_in[0];
    const float* scores = (const float*)d_in[1];
    const int*   labels = (const int*)d_in[2];

    const int N  = in_sizes[1];              // 200
    const int HW = in_sizes[0] / N;          // 243200 (divisible by 32)
    const int NT = (N + 15) / 16;            // 13
    const int NP = NT * 16;                  // 208

    // Workspace layout (floats): small arrays then the padded Gram matrix.
    float* ws          = (float*)d_ws;
    float* area        = ws;                 // 256
    int*   order       = (int*)(ws + 256);   // 256
    float* scores_s    = ws + 512;           // 256
    int*   labels_s    = (int*)(ws + 768);   // 256
    float* area_s      = ws + 1024;          // 256
    float* new_scores  = ws + 1280;          // 256
    int*   final_order = (int*)(ws + 1536);  // 256
    float* inter       = ws + 2048;          // NP*NP (~173 KB)

    // Output layout: scores(N) | labels(N) | masks(N*HW) | keep_inds(N)
    float* out        = (float*)d_out;
    float* out_scores = out;
    float* out_labels = out + N;
    float* out_masks  = out + 2 * (size_t)N;
    float* out_keep   = out + 2 * (size_t)N + (size_t)N * HW;

    area_kernel<<<N, 256, 0, stream>>>(masks, area, HW);
    sort1_kernel<<<1, 256, 0, stream>>>(scores, labels, area, order,
                                        scores_s, labels_s, area_s, N);
    dim3 g(NT, NT);
    gemm_kernel<<<g, 256, 0, stream>>>(masks, order, inter, N, HW, NP);
    decay_kernel<<<1, 256, 0, stream>>>(inter, area_s, labels_s, scores_s,
                                        new_scores, N, NP);
    sort2_kernel<<<1, 256, 0, stream>>>(new_scores, labels_s, order,
                                        out_scores, out_labels, out_keep,
                                        final_order, N);
    const int nv = HW >> 2;
    dim3 gc((nv + 255) / 256, N);
    copy_masks_kernel<<<gc, 256, 0, stream>>>(masks, final_order, out_masks, HW);
}
